// CifarMLP_34694745817434
// MI455X (gfx1250) — 8-bit matrix operand order repaired and checked by reading, compiled, not run on hardware
//
#include <hip/hip_runtime.h>
#include <hip/hip_bf16.h>
#include <stdint.h>


typedef __attribute__((ext_vector_type(8))) int v8i;

struct alignas(16) I4 { int a, b, c, d; };
union Frag { v8i v; long long q[4]; I4 x[2]; };

__device__ __forceinline__ signed char sgn8f(float v) {
    return (signed char)((v > 0.0f) - (v < 0.0f));
}

__device__ __forceinline__ unsigned lds_addr_of(const void* p) {
    return (unsigned)(unsigned long long)p;
}

__device__ __forceinline__ void async_load_b128(unsigned lds_dst,
                                                unsigned long long gaddr) {
    asm volatile("global_load_async_to_lds_b128 %0, %1, off"
                 :: "v"(lds_dst), "v"(gaddr) : "memory");
}

__device__ __forceinline__ void wait_async0() {
    asm volatile("s_wait_asynccnt 0" ::: "memory");
}

__global__ void k_binarize_pad(const float* __restrict__ W, int N, int K,
                               signed char* __restrict__ S, int Kp) {
    const int k = blockIdx.x * 256 + threadIdx.x;
    const int n = blockIdx.y;
    if (k >= Kp) return;
    signed char s = 0;
    if (n < N && k < K) s = sgn8f(W[(size_t)n * K + k]);
    S[(size_t)n * Kp + k] = s;
}

#define BM 128
#define BN 128
#define BKK 64
#define LSTR 80

__global__ __launch_bounds__(256)
void k_gemm_bin_big(const signed char* __restrict__ A,
                    const signed char* __restrict__ W,
                    short* __restrict__ Y,
                    int K, int ldy) {
    __shared__ __align__(16) signed char ldsA[2][BM * LSTR];
    __shared__ __align__(16) signed char ldsB[2][BN * LSTR];

    const int tid  = threadIdx.x;
    const int wid  = tid >> 5;
    const int lane = tid & 31;
    const int half = lane >> 4;
    const int l16  = lane & 15;
    const int wm   = wid >> 2;
    const int wn   = wid & 3;
    const int n0   = blockIdx.x * BN;
    const int m0   = blockIdx.y * BM;

    const int lrow = tid >> 1;
    const int lcol = (tid & 1) * 32;

    const unsigned long long gA =
        (unsigned long long)(A + (size_t)(m0 + lrow) * K + lcol);
    const unsigned long long gW =
        (unsigned long long)(W + (size_t)(n0 + lrow) * K + lcol);
    const unsigned la = lds_addr_of(&ldsA[0][lrow * LSTR + lcol]);
    const unsigned lb = lds_addr_of(&ldsB[0][lrow * LSTR + lcol]);
    const unsigned bufStepA = (unsigned)(BM * LSTR);
    const unsigned bufStepB = (unsigned)(BN * LSTR);

    v8i acc[4][2];
#pragma unroll
    for (int mi = 0; mi < 4; ++mi)
#pragma unroll
        for (int ni = 0; ni < 2; ++ni)
#pragma unroll
            for (int e = 0; e < 8; ++e) acc[mi][ni][e] = 0;

    async_load_b128(la,      gA);
    async_load_b128(la + 16, gA + 16);
    async_load_b128(lb,      gW);
    async_load_b128(lb + 16, gW + 16);

    int buf = 0;
    for (int kt = 0; kt < K; kt += BKK) {
        wait_async0();
        __syncthreads();

        const bool more = (kt + BKK) < K;
        if (more) {
            const int nb = buf ^ 1;
            const unsigned long long ga = gA + (unsigned)(kt + BKK);
            const unsigned long long gw = gW + (unsigned)(kt + BKK);
            async_load_b128(la + nb * bufStepA,      ga);
            async_load_b128(la + nb * bufStepA + 16, ga + 16);
            async_load_b128(lb + nb * bufStepB,      gw);
            async_load_b128(lb + nb * bufStepB + 16, gw + 16);
        }

        Frag fa[4], fb[2];
#pragma unroll
        for (int mi = 0; mi < 4; ++mi) {
            const signed char* p =
                &ldsA[buf][(wm * 64 + mi * 16 + l16) * LSTR + half * 16];
#pragma unroll
            for (int j = 0; j < 4; ++j)
                fa[mi].q[j] = *(const long long*)(p + (j & 1) * 8 + (j >> 1) * 32);
        }
#pragma unroll
        for (int ni = 0; ni < 2; ++ni) {
            const signed char* p =
                &ldsB[buf][(wn * 32 + ni * 16 + l16) * LSTR + half * 16];
            fb[ni].x[0] = *(const I4*)(p);
            fb[ni].x[1] = *(const I4*)(p + 32);
        }

#pragma unroll
        for (int mi = 0; mi < 4; ++mi)
#pragma unroll
            for (int ni = 0; ni < 2; ++ni)
                acc[mi][ni] = __builtin_amdgcn_wmma_i32_16x16x64_iu8(
                    true, fa[mi].v, true, fb[ni].v, acc[mi][ni], false, false);

        buf ^= 1;
    }

    short* yp = Y + (size_t)(m0 + wm * 64 + half * 8) * ldy
                  + (n0 + wn * 32 + l16);
#pragma unroll
    for (int mi = 0; mi < 4; ++mi) {
#pragma unroll
        for (int r = 0; r < 8; ++r) {
            short* rp = yp + (mi * 16 + r) * ldy;
            rp[0]  = (short)acc[mi][0][r];
            rp[16] = (short)acc[mi][1][r];
        }
    }
}

__global__ __launch_bounds__(256)
void k_gemm_bin_small(const signed char* __restrict__ A,
                      const signed char* __restrict__ W,
                      const float* __restrict__ bias,
                      void* __restrict__ out,
                      int K, int Nreal, int mode, int ldo) {
    const int tid  = threadIdx.x;
    const int wid  = tid >> 5;
    const int lane = tid & 31;
    const int half = lane >> 4;
    const int l16  = lane & 15;
    const int m0   = blockIdx.x * 128 + wid * 16;

    v8i acc;
#pragma unroll
    for (int e = 0; e < 8; ++e) acc[e] = 0;

    for (int kt = 0; kt < K; kt += 64) {
        Frag fa, fb;
        const signed char* pa = A + (size_t)(m0 + l16) * K + kt + half * 16;
#pragma unroll
        for (int j = 0; j < 4; ++j)
            fa.q[j] = *(const long long*)(pa + (j & 1) * 8 + (j >> 1) * 32);
        const signed char* pb = W + (size_t)l16 * K + kt + half * 16;
        fb.x[0] = *(const I4*)(pb);
        fb.x[1] = *(const I4*)(pb + 32);
        acc = __builtin_amdgcn_wmma_i32_16x16x64_iu8(
            true, fa.v, true, fb.v, acc, false, false);
    }

    const int mrow = m0 + half * 8;
    if (mode == 0) {
        short* yp = (short*)out + (size_t)mrow * ldo + l16;
#pragma unroll
        for (int r = 0; r < 8; ++r) yp[r * ldo] = (short)acc[r];
    } else {
        signed char* sp = (signed char*)out + (size_t)mrow * ldo + l16;
        const float bv = bias[l16];
#pragma unroll
        for (int r = 0; r < 8; ++r) {
            signed char s =
                (l16 < Nreal) ? sgn8f((float)acc[r] + bv) : (signed char)0;
            sp[r * ldo] = s;
            for (int c = 16; c < ldo; c += 16)
                sp[r * ldo + c] = 0;
        }
    }
}

__global__ void k_bn_stats_part(const short* __restrict__ Y, int M, int ldy,
                                int N, long long* __restrict__ ps1,
                                long long* __restrict__ ps2) {
    const int n = blockIdx.x * 256 + threadIdx.x;
    if (n >= N) return;
    const int chunk = M >> 6;
    const int mbeg  = blockIdx.y * chunk;
    long long a = 0, b = 0;
    const short* p = Y + (size_t)mbeg * ldy + n;
    for (int m = 0; m < chunk; ++m) {
        const int v = p[(size_t)m * ldy];
        a += v;
        b += v * v;
    }
    ps1[(size_t)n * 64 + blockIdx.y] = a;
    ps2[(size_t)n * 64 + blockIdx.y] = b;
}

__global__ void k_bn_finalize(const long long* __restrict__ ps1,
                              const long long* __restrict__ ps2,
                              int N, int M,
                              const float* __restrict__ g,
                              const float* __restrict__ be,
                              float* __restrict__ mean,
                              float* __restrict__ scale,
                              float* __restrict__ shift) {
    const int n = blockIdx.x * 256 + threadIdx.x;
    if (n >= N) return;
    long long a = 0, b = 0;
    for (int p = 0; p < 64; ++p) {
        a += ps1[(size_t)n * 64 + p];
        b += ps2[(size_t)n * 64 + p];
    }
    const double mu  = (double)a / (double)M;
    const double var = (double)b / (double)M - mu * mu;
    mean[n]  = (float)mu;
    scale[n] = rsqrtf((float)var + 1e-5f) * g[n];
    shift[n] = be[n];
}

__global__ void k_bn_binarize(const short* __restrict__ Y, int ldy,
                              const float* __restrict__ mean,
                              const float* __restrict__ scale,
                              const float* __restrict__ shift,
                              signed char* __restrict__ S, int ldS, int N) {
    const int n = blockIdx.x * 256 + threadIdx.x;
    const int m = blockIdx.y;
    if (n >= N) return;
    const float z =
        ((float)Y[(size_t)m * ldy + n] - mean[n]) * scale[n] + shift[n];
    S[(size_t)m * ldS + n] = sgn8f(z);
}

__global__ void k_bn_logsoftmax(const short* __restrict__ Y, int ldy,
                                const float* __restrict__ mean,
                                const float* __restrict__ scale,
                                const float* __restrict__ shift,
                                float* __restrict__ out, int M, int N) {
    const int m = blockIdx.x * blockDim.x + threadIdx.x;
    if (m >= M) return;
    float z[16];
    float mx = -3.4e38f;
    for (int n = 0; n < N; ++n) {
        const float v =
            ((float)Y[(size_t)m * ldy + n] - mean[n]) * scale[n] + shift[n];
        z[n] = v;
        mx = fmaxf(mx, v);
    }
    float se = 0.0f;
    for (int n = 0; n < N; ++n) se += __expf(z[n] - mx);
    const float lse = __logf(se) + mx;
    for (int n = 0; n < N; ++n) out[(size_t)m * N + n] = z[n] - lse;
}

extern "C" void kernel_launch(void* const* d_in, const int* in_sizes, int n_in,
                              void* d_out, int out_size, void* d_ws, size_t ws_size,
                              hipStream_t stream) {
    (void)in_sizes; (void)n_in; (void)out_size; (void)ws_size;
    const float* x  = (const float*)d_in[0];
    const float* w0 = (const float*)d_in[1];
    const float* w1 = (const float*)d_in[3];
    const float* w2 = (const float*)d_in[5];
    const float* w3 = (const float*)d_in[7];
    const float* b3 = (const float*)d_in[8];
    const float* w4 = (const float*)d_in[9];
    const float* b4 = (const float*)d_in[10];
    const float* w5 = (const float*)d_in[11];
    const float* b5 = (const float*)d_in[12];
    const float* g0 = (const float*)d_in[13]; const float* be0 = (const float*)d_in[14];
    const float* g1 = (const float*)d_in[15]; const float* be1 = (const float*)d_in[16];
    const float* g2 = (const float*)d_in[17]; const float* be2 = (const float*)d_in[18];
    const float* g3 = (const float*)d_in[19]; const float* be3 = (const float*)d_in[20];

    const int M = 16384;

    char* ws = (char*)d_ws;
    size_t off = 0;
    auto take = [&](size_t bytes) -> char* {
        char* p = ws + off;
        off += (bytes + 255) & ~(size_t)255;
        return p;
    };
    short*       y16 = (short*)take((size_t)M * 4096 * sizeof(short));
    signed char* sA  = (signed char*)take((size_t)M * 4096);
    signed char* sB  = (signed char*)take((size_t)M * 4096);
    signed char* sw0 = (signed char*)take((size_t)4096 * 3072);
    signed char* sw1 = (signed char*)take((size_t)4096 * 4096);
    signed char* sw2 = (signed char*)take((size_t)256 * 4096);
    signed char* sw3 = (signed char*)take((size_t)16 * 256);
    signed char* sw4 = (signed char*)take((size_t)16 * 64);
    signed char* sw5 = (signed char*)take((size_t)16 * 64);
    long long* ps1 = (long long*)take((size_t)4096 * 64 * sizeof(long long));
    long long* ps2 = (long long*)take((size_t)4096 * 64 * sizeof(long long));
    float* meanv  = (float*)take(4096 * sizeof(float));
    float* scalev = (float*)take(4096 * sizeof(float));
    float* shiftv = (float*)take(4096 * sizeof(float));

    const dim3 blk(256);
    auto cdiv = [](int a, int b) { return (a + b - 1) / b; };

    k_binarize_pad<<<dim3(cdiv(3072, 256), M),    blk, 0, stream>>>(x,  M,    3072, sA,  3072);
    k_binarize_pad<<<dim3(cdiv(3072, 256), 4096), blk, 0, stream>>>(w0, 4096, 3072, sw0, 3072);
    k_binarize_pad<<<dim3(cdiv(4096, 256), 4096), blk, 0, stream>>>(w1, 4096, 4096, sw1, 4096);
    k_binarize_pad<<<dim3(cdiv(4096, 256), 256),  blk, 0, stream>>>(w2, 256,  4096, sw2, 4096);
    k_binarize_pad<<<dim3(1, 16),                 blk, 0, stream>>>(w3, 16,   256,  sw3, 256);
    k_binarize_pad<<<dim3(1, 16),                 blk, 0, stream>>>(w4, 16,   16,   sw4, 64);
    k_binarize_pad<<<dim3(1, 16),                 blk, 0, stream>>>(w5, 10,   16,   sw5, 64);

    k_gemm_bin_big<<<dim3(4096 / BN, M / BM), blk, 0, stream>>>(sA, sw0, y16, 3072, 4096);
    k_bn_stats_part<<<dim3(16, 64), blk, 0, stream>>>(y16, M, 4096, 4096, ps1, ps2);
    k_bn_finalize<<<16, blk, 0, stream>>>(ps1, ps2, 4096, M, g0, be0, meanv, scalev, shiftv);
    k_bn_binarize<<<dim3(16, M), blk, 0, stream>>>(y16, 4096, meanv, scalev, shiftv, sB, 4096, 4096);

    k_gemm_bin_big<<<dim3(4096 / BN, M / BM), blk, 0, stream>>>(sB, sw1, y16, 4096, 4096);
    k_bn_stats_part<<<dim3(16, 64), blk, 0, stream>>>(y16, M, 4096, 4096, ps1, ps2);
    k_bn_finalize<<<16, blk, 0, stream>>>(ps1, ps2, 4096, M, g1, be1, meanv, scalev, shiftv);
    k_bn_binarize<<<dim3(16, M), blk, 0, stream>>>(y16, 4096, meanv, scalev, shiftv, sA, 4096, 4096);

    k_gemm_bin_big<<<dim3(256 / BN, M / BM), blk, 0, stream>>>(sA, sw2, y16, 4096, 256);
    k_bn_stats_part<<<dim3(1, 64), blk, 0, stream>>>(y16, M, 256, 256, ps1, ps2);
    k_bn_finalize<<<1, blk, 0, stream>>>(ps1, ps2, 256, M, g2, be2, meanv, scalev, shiftv);
    k_bn_binarize<<<dim3(1, M), blk, 0, stream>>>(y16, 256, meanv, scalev, shiftv, sB, 256, 256);

    k_gemm_bin_small<<<M / 128, blk, 0, stream>>>(sB, sw3, b3, sA, 256, 16, 1, 64);
    k_gemm_bin_small<<<M / 128, blk, 0, stream>>>(sA, sw4, b4, sB, 64, 16, 1, 64);
    k_gemm_bin_small<<<M / 128, blk, 0, stream>>>(sB, sw5, b5, y16, 64, 16, 0, 16);

    k_bn_stats_part<<<dim3(1, 64), blk, 0, stream>>>(y16, M, 16, 10, ps1, ps2);
    k_bn_finalize<<<1, blk, 0, stream>>>(ps1, ps2, 10, M, g3, be3, meanv, scalev, shiftv);
    k_bn_logsoftmax<<<cdiv(M, 256), blk, 0, stream>>>(
        y16, 16, meanv, scalev, shiftv, (float*)d_out, M, 10);
}
